// Block_31267361914924
// MI455X (gfx1250) — matrix operand order repaired and checked by reading, not compiled, not run on hardware
//
#include <hip/hip_runtime.h>

// ---------------------------------------------------------------------------
// Types for CDNA5 WMMA
// ---------------------------------------------------------------------------
typedef __attribute__((ext_vector_type(16))) __bf16 v16bf;
typedef __attribute__((ext_vector_type(8)))  float  v8f;

// Async global->LDS path (inline asm; CDNA5 VGLOBAL async ops + ASYNCcnt)
#if defined(__gfx1250__)
#define HAVE_ASYNC_LDS 1
#else
#define HAVE_ASYNC_LDS 0
#endif

__device__ __forceinline__ unsigned short f2bf(float f) {
    unsigned u = __float_as_uint(f);
    u += 0x7FFFu + ((u >> 16) & 1u);          // round-to-nearest-even
    return (unsigned short)(u >> 16);
}

// ---------------------------------------------------------------------------
// fp32 -> bf16 conversion (weights, once per launch)
// ---------------------------------------------------------------------------
__global__ void cvt_bf16_kernel(const float* __restrict__ in,
                                unsigned short* __restrict__ out, long long n) {
    long long i = (long long)blockIdx.x * blockDim.x + threadIdx.x;
    if (i < n) out[i] = f2bf(in[i]);
}

// ---------------------------------------------------------------------------
// LayerNorm over last dim (D=768), one block (256 threads) per row, bf16 out
// ---------------------------------------------------------------------------
__global__ void ln_kernel(const float* __restrict__ x,
                          const float* __restrict__ w,
                          const float* __restrict__ b,
                          unsigned short* __restrict__ y, int D) {
    __shared__ float rs1[256];
    __shared__ float rs2[256];
    const int row = blockIdx.x;
    const int tid = threadIdx.x;
    const float* xr = x + (long long)row * D;

    float s = 0.f, s2 = 0.f;
    for (int i = tid; i < D; i += 256) {
        float v = xr[i];
        s += v; s2 += v * v;
    }
    rs1[tid] = s; rs2[tid] = s2;
    __syncthreads();
    for (int off = 128; off > 0; off >>= 1) {
        if (tid < off) { rs1[tid] += rs1[tid + off]; rs2[tid] += rs2[tid + off]; }
        __syncthreads();
    }
    const float mu  = rs1[0] / (float)D;
    const float var = rs2[0] / (float)D - mu * mu;
    const float rstd = rsqrtf(var + 1e-6f);
    unsigned short* yr = y + (long long)row * D;
    for (int i = tid; i < D; i += 256)
        yr[i] = f2bf((xr[i] - mu) * rstd * w[i] + b[i]);
}

// ---------------------------------------------------------------------------
// Generic batched bf16 WMMA GEMM:  C = A(MxK) * B(KxN) [+bias][GELU][+res]
//   256 threads = 8 waves; block tile 128(M) x 64(N); K chunk 32.
//   Wave grid 4x2, each wave computes 32x32 = 2x2 v_wmma_f32_16x16x32_bf16.
//   REQUIREMENT: K % 32 == 0 (caller pads), lda/ldb give 16B-aligned rows.
//   Edge policy: staging CLAMPS source row/col (branchless); an OOB LDS
//   row/col only ever feeds an OOB C row/col, which the epilogue discards.
//   BT:       B source is stored (N x K) row-major (i.e. use B^T)
//   GELU:     exact gelu epilogue
//   OUT_BF16: store bf16 (else fp32)
// ---------------------------------------------------------------------------
template <bool BT, bool GELU, bool OUT_BF16>
__global__ void gemm_wmma_kernel(const unsigned short* __restrict__ A,
                                 const unsigned short* __restrict__ Bm,
                                 const float* __restrict__ bias,
                                 const float* __restrict__ residual,
                                 float* __restrict__ Cf,
                                 unsigned short* __restrict__ Cb,
                                 int M, int N, int K,
                                 int lda, int ldb, int ldc,
                                 long long aStride, long long bStride,
                                 int cHdiv, long long cs0, long long cs1) {
    __shared__ unsigned short ldsA[128 * 40];   // 128 rows x 32 k (stride 40)
    __shared__ unsigned short ldsB[64 * 40];    // 64 cols(n) x 32 k (stride 40)

    const int tid = threadIdx.x;
    const int bz  = blockIdx.z;
    A  += (long long)bz * aStride;
    Bm += (long long)bz * bStride;
    const long long cOff = (long long)(bz / cHdiv) * cs0 + (long long)(bz % cHdiv) * cs1;

    const int m0 = blockIdx.y * 128;
    const int n0 = blockIdx.x * 64;

    const int lane  = tid & 31;
    const int half  = lane >> 4;       // 0/1 lane group
    const int lm    = lane & 15;
    const int waveM = (tid >> 5) & 3;  // 32-row group
    const int waveN = tid >> 7;        // 32-col group

    v8f zero = {};
    v8f acc[2][2];
    acc[0][0] = zero; acc[0][1] = zero; acc[1][0] = zero; acc[1][1] = zero;

    // branchless clamped staging coordinates (computed once)
    const int arow0 = (tid + 0)   >> 2;            // 0..63
    const int arow1 = (tid + 256) >> 2;            // 64..127
    const int acol  = (tid & 3) * 8;               // 0/8/16/24
    const int agm0  = min(m0 + arow0, M - 1);
    const int agm1  = min(m0 + arow1, M - 1);

    for (int k0 = 0; k0 < K; k0 += 32) {
        // ---- stage A tile (128 x 32): 2 x 16B vectors per thread ----
#pragma unroll
        for (int j = 0; j < 2; ++j) {
            const int row = j ? arow1 : arow0;
            const int gm  = j ? agm1  : agm0;
            const unsigned short* gp = A + (long long)gm * lda + k0 + acol;
            uint4* lp = (uint4*)&ldsA[row * 40 + acol];
            if (k0 + 32 < K)                          // uniform scalar branch
                __builtin_prefetch(gp + 32, 0, 3);    // global_prefetch_b8
#if HAVE_ASYNC_LDS
            // async DMA: global -> LDS, tracked by ASYNCcnt
            {
                const unsigned ldsAddr =
                    (unsigned)(unsigned long long)(const void*)lp;
                asm volatile(
                    "global_load_async_to_lds_b128 %0, %1, off"
                    :
                    : "v"(ldsAddr), "v"((unsigned long long)gp)
                    : "memory");
            }
#else
            *lp = *(const uint4*)gp;
#endif
        }
        // ---- stage B tile (32 k x 64 n) into K-transposed LDS image ----
        if (BT) {
            // source is (N x K): one 16B vector along k per thread
            const int bn = tid >> 2;               // 0..63
            const int kv = (tid & 3) * 8;          // 0/8/16/24
            const int gn = min(n0 + bn, N - 1);    // clamp (epilogue discards)
            uint4 d = *(const uint4*)(Bm + (long long)gn * ldb + k0 + kv);
            *(uint4*)&ldsB[bn * 40 + kv] = d;
        } else {
            // source is (K x N): one 16B vector along n per thread, scatter k-major
            const int kk = tid >> 3;               // 0..31
            const int nv = (tid & 7) * 8;          // 0..56
            union { uint4 v; unsigned short s[8]; } u;
            u.v = *(const uint4*)(Bm + (long long)(k0 + kk) * ldb + n0 + nv);
#pragma unroll
            for (int i = 0; i < 8; ++i)
                ldsB[(nv + i) * 40 + kk] = u.s[i];
        }
#if HAVE_ASYNC_LDS
        asm volatile("s_wait_asynccnt 0x0" ::: "memory");
#endif
        __syncthreads();

        // ---- fragments (A 16x32 / B 32x16 bf16; this code's k order: k = 16*half + s in A and B) ----
        union Frag { v16bf v; unsigned u[8]; };
        Frag fa[2], fb[2];
#pragma unroll
        for (int mi = 0; mi < 2; ++mi) {
            const int rb = (waveM * 32 + mi * 16 + lm) * 40;
#pragma unroll
            for (int i = 0; i < 8; ++i) {
                const int kb = 2 * i + half * 16;
                fa[mi].u[i] = *(const unsigned*)&ldsA[rb + kb];
            }
        }
#pragma unroll
        for (int ni = 0; ni < 2; ++ni) {
            const int cb = (waveN * 32 + ni * 16 + lm) * 40 + half * 16;
#pragma unroll
            for (int i = 0; i < 8; ++i)
                fb[ni].u[i] = *(const unsigned*)&ldsB[cb + 2 * i];
        }
#pragma unroll
        for (int mi = 0; mi < 2; ++mi)
#pragma unroll
            for (int ni = 0; ni < 2; ++ni)
                acc[mi][ni] = __builtin_amdgcn_wmma_f32_16x16x32_bf16(
                    false, fa[mi].v, false, fb[ni].v,
                    (short)0, acc[mi][ni], false, false);

        __syncthreads();
    }

    // ---- epilogue ----
#pragma unroll
    for (int mi = 0; mi < 2; ++mi) {
#pragma unroll
        for (int ni = 0; ni < 2; ++ni) {
            const int nn = n0 + waveN * 32 + ni * 16 + lm;
            if (nn >= N) continue;
            const float bi = bias ? bias[nn] : 0.0f;
#pragma unroll
            for (int r = 0; r < 8; ++r) {
                const int mm = m0 + waveM * 32 + mi * 16 + half * 8 + r;
                if (mm >= M) continue;
                float v = acc[mi][ni][r] + bi;
                if (GELU)
                    v = 0.5f * v * (1.0f + erff(v * 0.70710678118654752f));
                const long long ci = cOff + (long long)mm * ldc + nn;
                if (residual) v += residual[ci];
                if (OUT_BF16) Cb[ci] = f2bf(v);
                else          Cf[ci] = v;
            }
        }
    }
}

// ---------------------------------------------------------------------------
// qkv (M x 2304 fp32, bias applied) -> per-head bf16 q/k (B,h,T,dh) and
// v (B,h,TP,dh) with TP-padded rows (pad pre-zeroed). Scale folded into q.
// ---------------------------------------------------------------------------
__global__ void repack_qkv_kernel(const float* __restrict__ qkv,
                                  unsigned short* __restrict__ q,
                                  unsigned short* __restrict__ k,
                                  unsigned short* __restrict__ v,
                                  int Mtot, int T, int TP) {
    const long long idx = (long long)blockIdx.x * blockDim.x + threadIdx.x;
    const long long total = (long long)Mtot * 2304;
    if (idx >= total) return;
    const int m = (int)(idx / 2304);
    const int c = (int)(idx % 2304);
    const int which = c / 768;
    const int rem   = c % 768;
    const int h = rem / 64;
    const int d = rem % 64;
    const int b = m / T;
    const int t = m % T;
    const float val = qkv[idx];
    const int bh = b * 12 + h;
    if (which == 0)
        q[(((long long)bh) * T + t) * 64 + d] = f2bf(val * 0.125f);  // 64^-0.5
    else if (which == 1)
        k[(((long long)bh) * T + t) * 64 + d] = f2bf(val);
    else
        v[(((long long)bh) * TP + t) * 64 + d] = f2bf(val);
}

// ---------------------------------------------------------------------------
// softmax over the HEADS axis (faithful: axis=1 of (B,h,T,T)).
// One thread per (b, q, kcol) with kcol < TP; pad columns get zeros so the
// downstream GEMM can run K=TP unguarded.
// ---------------------------------------------------------------------------
__global__ void softmax_heads_kernel(const float* __restrict__ dp,
                                     unsigned short* __restrict__ attn,
                                     int Bn, int T, int TP) {
    const long long idx = (long long)blockIdx.x * blockDim.x + threadIdx.x;
    const long long total = (long long)Bn * T * TP;
    if (idx >= total) return;
    const int kcol = (int)(idx % TP);
    const long long bq = idx / TP;
    const int q = (int)(bq % T);
    const int b = (int)(bq / T);
    const long long hs   = (long long)T * TP;              // head stride
    const long long base = ((long long)b * 12 * T + q) * TP + kcol;
    if (kcol >= T) {
#pragma unroll
        for (int h = 0; h < 12; ++h) attn[base + (long long)h * hs] = 0;
        return;
    }
    float vals[12];
    float mx = -3.402823466e38f;
#pragma unroll
    for (int h = 0; h < 12; ++h) {
        vals[h] = dp[base + (long long)h * hs];
        mx = fmaxf(mx, vals[h]);
    }
    float s = 0.f;
#pragma unroll
    for (int h = 0; h < 12; ++h) { vals[h] = expf(vals[h] - mx); s += vals[h]; }
    const float inv = 1.0f / s;
#pragma unroll
    for (int h = 0; h < 12; ++h)
        attn[base + (long long)h * hs] = f2bf(vals[h] * inv);
}

// ---------------------------------------------------------------------------
// Host-side orchestration
// ---------------------------------------------------------------------------
static inline long long cdiv(long long a, long long b) { return (a + b - 1) / b; }

extern "C" void kernel_launch(void* const* d_in, const int* in_sizes, int n_in,
                              void* d_out, int out_size, void* d_ws, size_t ws_size,
                              hipStream_t stream) {
    (void)in_sizes; (void)n_in; (void)out_size; (void)ws_size;

    const float* x      = (const float*)d_in[0];
    const float* ln1_w  = (const float*)d_in[1];
    const float* ln1_b  = (const float*)d_in[2];
    const float* qkv_w  = (const float*)d_in[3];
    const float* qkv_b  = (const float*)d_in[4];
    const float* proj_w = (const float*)d_in[5];
    const float* proj_b = (const float*)d_in[6];
    const float* ln2_w  = (const float*)d_in[7];
    const float* ln2_b  = (const float*)d_in[8];
    const float* fc1_w  = (const float*)d_in[9];
    const float* fc1_b  = (const float*)d_in[10];
    const float* fc2_w  = (const float*)d_in[11];
    const float* fc2_b  = (const float*)d_in[12];
    float* out = (float*)d_out;

    const int Bn = 32, T = 577, D = 768, NH = 12, DH = 64;
    const int TP = 608;               // T padded to multiple of 32
    const int M  = Bn * T;            // 18464
    const int N3 = 3 * D;             // 2304
    const int HM = 4 * D;             // 3072

    // ---- workspace carve-out ----
    char* ws = (char*)d_ws;
    size_t off = 0;
    auto carve = [&](size_t bytes) -> void* {
        off = (off + 255) & ~(size_t)255;
        void* p = ws + off;
        off += bytes;
        return p;
    };
    unsigned short* wqkv  = (unsigned short*)carve((size_t)D * N3 * 2);
    unsigned short* wproj = (unsigned short*)carve((size_t)D * D * 2);
    unsigned short* wfc1  = (unsigned short*)carve((size_t)D * HM * 2);
    unsigned short* wfc2  = (unsigned short*)carve((size_t)HM * D * 2);
    unsigned short* yln   = (unsigned short*)carve((size_t)M * D * 2);
    float*          qkv   = (float*)carve((size_t)M * N3 * 4);
    unsigned short* qb    = (unsigned short*)carve((size_t)Bn * NH * T  * DH * 2);
    unsigned short* kb    = (unsigned short*)carve((size_t)Bn * NH * T  * DH * 2);
    unsigned short* vb    = (unsigned short*)carve((size_t)Bn * NH * TP * DH * 2);
    float*          sc    = (float*)carve((size_t)Bn * NH * T * TP * 4);
    unsigned short* attn  = (unsigned short*)carve((size_t)Bn * NH * T * TP * 2);
    unsigned short* wa    = (unsigned short*)carve((size_t)M * D * 2);
    float*          x1    = (float*)carve((size_t)M * D * 4);
    unsigned short* y2    = (unsigned short*)carve((size_t)M * D * 2);
    unsigned short* hmlp  = (unsigned short*)carve((size_t)M * HM * 2);

    // ---- 1. convert weights to bf16 ----
    {
        long long n;
        n = (long long)D * N3;
        cvt_bf16_kernel<<<cdiv(n,256),256,0,stream>>>(qkv_w,  wqkv,  n);
        n = (long long)D * D;
        cvt_bf16_kernel<<<cdiv(n,256),256,0,stream>>>(proj_w, wproj, n);
        n = (long long)D * HM;
        cvt_bf16_kernel<<<cdiv(n,256),256,0,stream>>>(fc1_w,  wfc1,  n);
        n = (long long)HM * D;
        cvt_bf16_kernel<<<cdiv(n,256),256,0,stream>>>(fc2_w,  wfc2,  n);
    }

    // ---- 2. LN1 ----
    ln_kernel<<<M,256,0,stream>>>(x, ln1_w, ln1_b, yln, D);

    // ---- 3. qkv GEMM: (M x 768) @ (768 x 2304) + bias -> fp32 ----
    {
        dim3 g((unsigned)cdiv(N3,64), (unsigned)cdiv(M,128), 1);
        gemm_wmma_kernel<false,false,false><<<g,256,0,stream>>>(
            yln, wqkv, qkv_b, nullptr, qkv, nullptr,
            M, N3, D, D, N3, N3, 0, 0, 1, 0, 0);
    }

    // ---- 4. zero v pad rows, then repack q/k/v per head ----
    (void)hipMemsetAsync(vb, 0, (size_t)Bn * NH * TP * DH * 2, stream);
    {
        long long n = (long long)M * N3;
        repack_qkv_kernel<<<cdiv(n,256),256,0,stream>>>(qkv, qb, kb, vb, M, T, TP);
    }

    // ---- 5. scores: per (b,h)  q(T x 64) @ k^T(64 x T) -> fp32 (ldc=TP) ----
    {
        dim3 g((unsigned)cdiv(T,64), (unsigned)cdiv(T,128), Bn*NH);
        gemm_wmma_kernel<true,false,false><<<g,256,0,stream>>>(
            qb, kb, nullptr, nullptr, sc, nullptr,
            T, T, DH, DH, DH, TP,
            (long long)T*DH, (long long)T*DH, 1, (long long)T*TP, 0);
    }

    // ---- 6. softmax over heads axis -> bf16 attn (pad cols zeroed) ----
    {
        long long n = (long long)Bn * T * TP;
        softmax_heads_kernel<<<cdiv(n,256),256,0,stream>>>(sc, attn, Bn, T, TP);
    }

    // ---- 7. attn @ v: per (b,h) (T x TP)@(TP x 64) -> wa bf16 in (B,T,D) ----
    {
        dim3 g(1, (unsigned)cdiv(T,128), Bn*NH);
        gemm_wmma_kernel<false,false,true><<<g,256,0,stream>>>(
            attn, vb, nullptr, nullptr, nullptr, wa,
            T, DH, TP, TP, DH, D,
            (long long)T*TP, (long long)TP*DH, NH, (long long)T*D, (long long)DH);
    }

    // ---- 8. proj GEMM + residual(x) -> x1 fp32 ----
    {
        dim3 g((unsigned)cdiv(D,64), (unsigned)cdiv(M,128), 1);
        gemm_wmma_kernel<false,false,false><<<g,256,0,stream>>>(
            wa, wproj, proj_b, x, x1, nullptr,
            M, D, D, D, D, D, 0, 0, 1, 0, 0);
    }

    // ---- 9. LN2 ----
    ln_kernel<<<M,256,0,stream>>>(x1, ln2_w, ln2_b, y2, D);

    // ---- 10. fc1 GEMM + exact GELU -> bf16 hidden ----
    {
        dim3 g((unsigned)cdiv(HM,64), (unsigned)cdiv(M,128), 1);
        gemm_wmma_kernel<false,true,true><<<g,256,0,stream>>>(
            y2, wfc1, fc1_b, nullptr, nullptr, hmlp,
            M, HM, D, D, HM, HM, 0, 0, 1, 0, 0);
    }

    // ---- 11. fc2 GEMM + residual(x1) -> out fp32 ----
    {
        dim3 g((unsigned)cdiv(D,64), (unsigned)cdiv(M,128), 1);
        gemm_wmma_kernel<false,false,false><<<g,256,0,stream>>>(
            hmlp, wfc2, fc2_b, x1, out, nullptr,
            M, D, HM, HM, D, D, 0, 0, 1, 0, 0);
    }
}
